// ExpansionContrastModule_7352984011481
// MI455X (gfx1250) — matrix operand order repaired and checked by reading, not compiled, not run on hardware
//
#include <hip/hip_runtime.h>
#include <hip/hip_bf16.h>

typedef __attribute__((ext_vector_type(16))) _Float16 v16h;
typedef __attribute__((ext_vector_type(8)))  float    v8f;

#define CSWAP(a, b) { float _lo = fminf(a, b); float _hi = fmaxf(a, b); (a) = _lo; (b) = _hi; }

// Plane = one (batch, channel) 256x256 image; PLANE_ELEMS = 65536.
// x / y / branch buffers all live in d_ws as f32, layout (b, c, h, w).
static constexpr int PLANE_ELEMS = 65536;          // 256*256
static constexpr size_t BUF_ELEMS = 8ull * 16 * PLANE_ELEMS; // 8.39M floats per buffer

// ------------------------------------------------------------------
// Kernel 1: in_conv 1x1, 64 -> 16 channels, via v_wmma_f32_16x16x32_f16.
// One wave owns a 16-pixel x 16-outchannel tile; K=64 in two WMMA steps.
// ------------------------------------------------------------------
__global__ void inconv_wmma_kernel(const float* __restrict__ cen,   // (8,64,256,256)
                                   const float* __restrict__ w,     // (16 oc, 64 ic)
                                   const float* __restrict__ bias,  // (16,)
                                   float* __restrict__ x)           // (8,16,256,256)
{
    __shared__ float lds[8 * 16 * 17];     // 8 waves * 16x16 tile (pad 17)
    const int lane = threadIdx.x & 31;
    const int wv   = threadIdx.x >> 5;
    const int group = blockIdx.x * 8 + wv;           // 32768 tiles total
    const int b = group >> 12;                       // 4096 tiles per batch
    const int pixbase = (group & 4095) << 4;         // 16 consecutive pixels
    const int m  = lane & 15;                        // A: row M / B: col N (=oc)
    const int hi = lane >> 4;

    v8f acc = {};
    #pragma unroll
    for (int kb = 0; kb < 64; kb += 32) {
        v16h a, bf;
        // A-matrix (16x32 f16): lanes 0-15 hold K=0..7,16..23; lanes 16-31 hold K=8..15,24..31
        #pragma unroll
        for (int e = 0; e < 16; ++e) {
            const int K = (e < 8 ? e : e + 8) + hi * 8;
            a[e] = (_Float16)cen[((b * 64 + kb + K) << 16) + pixbase + m];
        }
        // B-matrix (32x16 f16): lanes 0-15 hold K=0..7,16..23; lanes 16-31 hold K=8..15,24..31
        #pragma unroll
        for (int e = 0; e < 16; ++e) {
            const int K = (e < 8 ? e : e + 8) + hi * 8;
            bf[e] = (_Float16)w[m * 64 + kb + K];
        }
        acc = __builtin_amdgcn_wmma_f32_16x16x32_f16(false, a, false, bf,
                                                     (short)0, acc, false, false);
    }

    // D layout: lane l, VGPR v -> M = v + 8*(l>=16), N = l&15. Add bias[N], LDS transpose.
    const float bb = bias[m];
    float* L = lds + wv * (16 * 17);
    #pragma unroll
    for (int v = 0; v < 8; ++v)
        L[(v + hi * 8) * 17 + m] = acc[v] + bb;
    __syncthreads();

    // Coalesced store: lane -> (oc = lane>>1, 8-pixel half), two float4 stores.
    const int oc = lane >> 1;
    const int ph = (lane & 1) * 8;
    float4 u0, u1;
    u0.x = L[(ph + 0) * 17 + oc]; u0.y = L[(ph + 1) * 17 + oc];
    u0.z = L[(ph + 2) * 17 + oc]; u0.w = L[(ph + 3) * 17 + oc];
    u1.x = L[(ph + 4) * 17 + oc]; u1.y = L[(ph + 5) * 17 + oc];
    u1.z = L[(ph + 6) * 17 + oc]; u1.w = L[(ph + 7) * 17 + oc];
    float* xp = x + ((b * 16 + oc) << 16) + pixbase + ph;
    reinterpret_cast<float4*>(xp)[0] = u0;
    reinterpret_cast<float4*>(xp)[1] = u1;
}

// ------------------------------------------------------------------
// Kernel 2a: depthwise SxS conv (zero pad S/2). One (b,c) plane per
// block slice -> channel index is block-uniform -> scalar weight loads.
// grid = 128 planes * 256 tiles, block = 256.
// ------------------------------------------------------------------
template <int S>
__global__ void dwconv_kernel(const float* __restrict__ x,
                              const float* __restrict__ w,    // (16,1,S,S)
                              const float* __restrict__ bias, // (16,)
                              float* __restrict__ y)
{
    constexpr int P = S / 2;
    const int plane = blockIdx.x >> 8;          // (b*16 + c)
    const int tile  = blockIdx.x & 255;
    const int c = plane & 15;
    const int p = tile * 256 + threadIdx.x;
    const int h = p >> 8, wc = p & 255;
    const float* xp = x + plane * PLANE_ELEMS;

    float acc = bias[c];
    #pragma unroll
    for (int i = 0; i < S; ++i) {
        const int hh = h + i - P;
        if (hh < 0 || hh >= 256) continue;
        #pragma unroll
        for (int j = 0; j < S; ++j) {
            const int ww = wc + j - P;
            if (ww < 0 || ww >= 256) continue;
            acc = fmaf(w[(c * S + i) * S + j], xp[hh * 256 + ww], acc);
        }
    }
    y[plane * PLANE_ELEMS + p] = acc;
}

// ------------------------------------------------------------------
// Kernel 2b: circ-shift branch body. 8 dilated neighbor diffs, grouped
// 1x1 (4->1 per channel), *T, sort-8 (Batcher, 19 comparators),
// grouped 1x1 (8->1 per channel).
// ------------------------------------------------------------------
template <int S>
__global__ void branch_kernel(const float* __restrict__ y,
                              const float* __restrict__ w1,  // (16,4) for this branch
                              const float* __restrict__ b1,  // (16,)
                              const float* __restrict__ w2,  // (16,8)
                              const float* __restrict__ b2,  // (16,)
                              float* __restrict__ bo)        // branch output buffer
{
    const int plane = blockIdx.x >> 8;
    const int tile  = blockIdx.x & 255;
    const int c = plane & 15;
    const int p = tile * 256 + threadIdx.x;
    const int h = p >> 8, wc = p & 255;
    const float* yp = y + plane * PLANE_ELEMS;

    const float yc = yp[p];
    const int dy[8] = {-1, -1, -1, 0, 1, 1, 1, 0};
    const int dx[8] = {-1, 0, 1, 1, 1, 0, -1, -1};

    float T[8];
    #pragma unroll
    for (int k = 0; k < 8; ++k) {
        const int hh = h + dy[k] * S;
        const int ww = wc + dx[k] * S;
        float nv = 0.f;
        if (hh >= 0 && hh < 256 && ww >= 0 && ww < 256) nv = yp[hh * 256 + ww];
        T[k] = yc - nv;
    }
    float Sv[4];
    #pragma unroll
    for (int k = 0; k < 4; ++k) Sv[k] = T[k] + T[k + 4];

    const float w10 = w1[c * 4 + 0], w11 = w1[c * 4 + 1];
    const float w12 = w1[c * 4 + 2], w13 = w1[c * 4 + 3];
    const float b1c = b1[c];

    float o[8];
    #pragma unroll
    for (int k = 0; k < 8; ++k) {
        float v = b1c;
        v = fmaf(w10, Sv[(k + 1) & 3], v);
        v = fmaf(w11, Sv[(k + 3) & 3], v);
        v = fmaf(w12, Sv[(k + 2) & 3], v);
        v = fmaf(w13, 2.f * T[(k + 4) & 7], v);
        o[k] = v * T[k];
    }

    // Batcher odd-even mergesort, 8 elements, ascending.
    CSWAP(o[0], o[1]); CSWAP(o[2], o[3]); CSWAP(o[4], o[5]); CSWAP(o[6], o[7]);
    CSWAP(o[0], o[2]); CSWAP(o[1], o[3]); CSWAP(o[4], o[6]); CSWAP(o[5], o[7]);
    CSWAP(o[1], o[2]); CSWAP(o[5], o[6]); CSWAP(o[0], o[4]); CSWAP(o[3], o[7]);
    CSWAP(o[1], o[5]); CSWAP(o[2], o[6]);
    CSWAP(o[1], o[4]); CSWAP(o[3], o[6]);
    CSWAP(o[2], o[4]); CSWAP(o[3], o[5]);
    CSWAP(o[3], o[4]);

    float r = b2[c];
    #pragma unroll
    for (int k = 0; k < 8; ++k) r = fmaf(o[k], w2[c * 8 + k], r);
    bo[plane * PLANE_ELEMS + p] = r;
}

// ------------------------------------------------------------------
// Kernel 3: sort-4 over branches, base_w contraction, BN + SiLU,
// 16->1 final conv, sigmoid.
// ------------------------------------------------------------------
__global__ void final_kernel(const float* __restrict__ bo,       // 4 branch buffers
                             const float* __restrict__ base_w,   // (16,4)
                             const float* __restrict__ bn_s,     // (16,)
                             const float* __restrict__ bn_b,     // (16,)
                             const float* __restrict__ fin_w,    // (1,16)
                             const float* __restrict__ fin_b,    // (1,)
                             float* __restrict__ out)            // (8,1,256,256)
{
    const int idx = blockIdx.x * 256 + threadIdx.x;  // b*65536 + p
    const int b = idx >> 16;
    const int p = idx & 65535;

    float acc = 0.f;
    #pragma unroll
    for (int c = 0; c < 16; ++c) {
        const size_t base = (size_t)(b * 16 + c) * PLANE_ELEMS + p;
        float a0 = bo[base];
        float a1 = bo[base + BUF_ELEMS];
        float a2 = bo[base + 2 * BUF_ELEMS];
        float a3 = bo[base + 3 * BUF_ELEMS];
        CSWAP(a0, a1); CSWAP(a2, a3); CSWAP(a0, a2); CSWAP(a1, a3); CSWAP(a1, a2);
        float v = a0 * base_w[c * 4 + 0];
        v = fmaf(a1, base_w[c * 4 + 1], v);
        v = fmaf(a2, base_w[c * 4 + 2], v);
        v = fmaf(a3, base_w[c * 4 + 3], v);
        const float t = fmaf(v, bn_s[c], bn_b[c]);
        const float act = t / (1.f + __expf(-t));      // SiLU
        acc = fmaf(act, fin_w[c], acc);
    }
    acc += fin_b[0];
    out[idx] = 1.f / (1.f + __expf(-acc));
}

// ------------------------------------------------------------------
extern "C" void kernel_launch(void* const* d_in, const int* in_sizes, int n_in,
                              void* d_out, int out_size, void* d_ws, size_t ws_size,
                              hipStream_t stream) {
    (void)in_sizes; (void)n_in; (void)out_size; (void)ws_size;
    const float* cen      = (const float*)d_in[0];
    // d_in[1] = mas (unused by the reference)
    const float* in_w     = (const float*)d_in[2];
    const float* in_b     = (const float*)d_in[3];
    const float* dw_w[4]  = {(const float*)d_in[4], (const float*)d_in[6],
                             (const float*)d_in[8], (const float*)d_in[10]};
    const float* dw_b[4]  = {(const float*)d_in[5], (const float*)d_in[7],
                             (const float*)d_in[9], (const float*)d_in[11]};
    const float* l1_w     = (const float*)d_in[12];  // (4,16,4)
    const float* l1_b     = (const float*)d_in[13];  // (4,16)
    const float* l2_w     = (const float*)d_in[14];  // (4,16,8)
    const float* l2_b     = (const float*)d_in[15];  // (4,16)
    const float* base_w   = (const float*)d_in[16];
    const float* bn_s     = (const float*)d_in[17];
    const float* bn_b     = (const float*)d_in[18];
    const float* fin_w    = (const float*)d_in[19];
    const float* fin_b    = (const float*)d_in[20];

    float* ws = (float*)d_ws;
    float* x  = ws;                      // 33.5 MB: in_conv output
    float* y  = ws + BUF_ELEMS;          // 33.5 MB: per-branch depthwise output
    float* bo = ws + 2 * BUF_ELEMS;      // 4 x 33.5 MB: branch outputs

    // 1) 1x1 conv 64->16 on the matrix pipe (f16 in, f32 accumulate)
    inconv_wmma_kernel<<<4096, 256, 0, stream>>>(cen, in_w, in_b, x);

    // 2) four branches: depthwise conv then branch body
    dwconv_kernel<1><<<32768, 256, 0, stream>>>(x, dw_w[0], dw_b[0], y);
    branch_kernel<1><<<32768, 256, 0, stream>>>(y, l1_w + 0 * 64, l1_b + 0 * 16,
                                                l2_w + 0 * 128, l2_b + 0 * 16,
                                                bo + 0 * BUF_ELEMS);
    dwconv_kernel<3><<<32768, 256, 0, stream>>>(x, dw_w[1], dw_b[1], y);
    branch_kernel<3><<<32768, 256, 0, stream>>>(y, l1_w + 1 * 64, l1_b + 1 * 16,
                                                l2_w + 1 * 128, l2_b + 1 * 16,
                                                bo + 1 * BUF_ELEMS);
    dwconv_kernel<5><<<32768, 256, 0, stream>>>(x, dw_w[2], dw_b[2], y);
    branch_kernel<5><<<32768, 256, 0, stream>>>(y, l1_w + 2 * 64, l1_b + 2 * 16,
                                                l2_w + 2 * 128, l2_b + 2 * 16,
                                                bo + 2 * BUF_ELEMS);
    dwconv_kernel<7><<<32768, 256, 0, stream>>>(x, dw_w[3], dw_b[3], y);
    branch_kernel<7><<<32768, 256, 0, stream>>>(y, l1_w + 3 * 64, l1_b + 3 * 16,
                                                l2_w + 3 * 128, l2_b + 3 * 16,
                                                bo + 3 * BUF_ELEMS);

    // 3) combine: sort-4, base conv, BN+SiLU, 16->1, sigmoid
    final_kernel<<<2048, 256, 0, stream>>>(bo, base_w, bn_s, bn_b, fin_w, fin_b,
                                           (float*)d_out);
}
